// Model_55671366091256
// MI455X (gfx1250) — compile-verified, matrix operand order checked by reading, not run on hardware
//
#include <hip/hip_runtime.h>
#include <math.h>

// ---------------------------------------------------------------------------
// Problem constants (from the reference)
// ---------------------------------------------------------------------------
constexpr int Dm   = 512;
constexpr int HIDm = 2048;
constexpr int Lm   = 4;
constexpr int Vm   = 32000;
constexpr int Bm   = 2;
constexpr int Tm   = 1025;
constexpr int MBm  = 128;
constexpr int NHm  = 8;
constexpr int DHm  = 64;           // Dm / NHm
constexpr int H3   = 3 * Dm;       // 1536
constexpr int Nrow = Bm * MBm;     // 256 rows per chunk
constexpr float LRc = 0.0003f;

typedef unsigned int u32;
typedef int v4i __attribute__((__vector_size__(16)));   // int4 vector for async-LDS builtin

// ---------------------------------------------------------------------------
// CDNA5 feature detection (async global->LDS copies, gfx1250)
// ---------------------------------------------------------------------------
#if defined(__HIP_DEVICE_COMPILE__) && defined(__has_builtin)
#  if __has_builtin(__builtin_amdgcn_global_load_async_to_lds_b128)
#    define HAVE_ASYNC_LDS 1
#  endif
#endif
#ifndef HAVE_ASYNC_LDS
#  define HAVE_ASYNC_LDS 0
#endif

#if defined(__HIP_DEVICE_COMPILE__)
#  if defined(__has_builtin) && __has_builtin(__builtin_amdgcn_s_wait_asynccnt)
#    define ASYNC_WAIT() __builtin_amdgcn_s_wait_asynccnt(0)
#  else
#    define ASYNC_WAIT() asm volatile("s_wait_asynccnt 0" ::: "memory")
#  endif
#else
#  define ASYNC_WAIT() ((void)0)
#endif

#if HAVE_ASYNC_LDS
#  define AS_GLOBAL(p) ((__attribute__((address_space(1))) v4i*)(p))
#  define AS_LDS(p)    ((__attribute__((address_space(3))) v4i*)(p))
#endif

// ---------------------------------------------------------------------------
// CDNA5 WMMA types (wave32, v_wmma_f32_16x16x32_bf16)
// ---------------------------------------------------------------------------
typedef __attribute__((ext_vector_type(16))) __bf16 bf16x16;
typedef __attribute__((ext_vector_type(8)))  __bf16 bf16x8;
typedef __attribute__((ext_vector_type(8)))  float  f32x8;

// GEMM tiling: block = 64(M) x 128(N) x 32(K), 256 threads = 8 wave32 waves,
// each wave owns a 32x32 quadrant (2x2 WMMA 16x16 tiles).
constexpr int BM = 64;
constexpr int BN = 128;
constexpr int BK = 32;
constexpr int SA = BK + 8;   // 80-byte LDS row stride: 16B aligned, 20-dword
                             // stagger -> conflict-free b128 fragment loads

// A operand: fp32 activations (paired b64 loads -> packed bf16, reg-pipelined).
// W operand: bf16.
// MODE 1 (NT): C[M,N] = A[M,K] @ W^T, W is [N,K] row-major.
//   Used for ALL forward GEMMs (weights pre-transposed once per call) and all
//   backward dX = dY @ W^T GEMMs. B tile is contiguous 64B runs -> staged with
//   CDNA5 async global->LDS DMA into a double-buffered LDS tile (ASYNCcnt).
// MODE 2 (TN): C[M,N] = A^T @ W,  A is [K,M], W is [K,N]  (dW = X^T @ dY)
template <int MODE>
__global__ __launch_bounds__(256, 2)
void gemm_wmma(const float* __restrict__ A, const __bf16* __restrict__ W,
               float* __restrict__ C, int M, int N, int K,
               const float* __restrict__ bias, const float* __restrict__ residual,
               int kChunk)
{
    constexpr int NBUF = (MODE == 1 && HAVE_ASYNC_LDS) ? 2 : 1;
    __shared__ __attribute__((aligned(16))) __bf16 As[BM * SA];
    __shared__ __attribute__((aligned(16))) __bf16 Bsd[NBUF * BN * SA];

    const int tid  = threadIdx.x;
    const int m0   = blockIdx.y * BM;
    const int n0   = blockIdx.x * BN;
    int kBeg = blockIdx.z * kChunk;
    int kEnd = kBeg + kChunk; if (kEnd > K) kEnd = K;

    const int wave = tid >> 5;      // wave32
    const int lane = tid & 31;
    const int half = lane >> 4;     // K-half selector per ISA 16-bit layouts
    const int l16  = lane & 15;
    const int wm   = (wave & 1) * 32;
    const int wn   = (wave >> 1) * 32;

    f32x8 acc[2][2] = {};

    // ---- A staging: fp32 pairs -> packed bf16, register-pipelined ----
    float2 aR[4];                       // 1024 fp32-pairs / 256 threads
    auto loadA = [&](int k0) {
#pragma unroll
        for (int j = 0; j < 4; ++j) {
            int q = tid + j * 256;
            if (MODE == 2) { int kk = q >> 5, mp = (q & 31) << 1;
                             aR[j] = *(const float2*)(A + (size_t)(k0 + kk) * M + (m0 + mp)); }
            else           { int mm = q >> 4, kp = (q & 15) << 1;
                             aR[j] = *(const float2*)(A + (size_t)(m0 + mm) * K + (k0 + kp)); }
        }
    };
    auto storeA = [&]() {
#pragma unroll
        for (int j = 0; j < 4; ++j) {
            int q = tid + j * 256;
            if (MODE == 2) {            // transpose: two b16 stores to adjacent rows
                int kk = q >> 5, mp = (q & 31) << 1;
                As[(mp    ) * SA + kk] = (__bf16)aR[j].x;
                As[(mp + 1) * SA + kk] = (__bf16)aR[j].y;
            } else {                    // packed b32 store (aligned: kp even)
                int mm = q >> 4, kp = (q & 15) << 1;
                union { __bf16 h[2]; u32 w; } pk;
                pk.h[0] = (__bf16)aR[j].x;
                pk.h[1] = (__bf16)aR[j].y;
                *(u32*)&As[mm * SA + kp] = pk.w;
            }
        }
    };

    // ---- B staging ----
#if HAVE_ASYNC_LDS
    // MODE 1: async DMA straight into LDS (double-buffered); 512 16B segments.
    auto issueB = [&](int k0, int p) {
#pragma unroll
        for (int j = 0; j < 2; ++j) {
            int idx = tid + j * 256;
            int nn = idx >> 2, seg = idx & 3;
            const __bf16* gp = W + (size_t)(n0 + nn) * K + k0 + seg * 8;
            __builtin_amdgcn_global_load_async_to_lds_b128(
                AS_GLOBAL(gp), AS_LDS(&Bsd[p * BN * SA + nn * SA + seg * 8]), 0, 0);
        }
    };
#else
    uint4 bR4[2];
    auto loadB1 = [&](int k0) {
#pragma unroll
        for (int j = 0; j < 2; ++j) {
            int idx = tid + j * 256;
            int nn = idx >> 2, seg = idx & 3;
            bR4[j] = *(const uint4*)(W + (size_t)(n0 + nn) * K + k0 + seg * 8);
        }
    };
    auto storeB1 = [&]() {
#pragma unroll
        for (int j = 0; j < 2; ++j) {
            int idx = tid + j * 256;
            int nn = idx >> 2, seg = idx & 3;
            *(uint4*)&Bsd[nn * SA + seg * 8] = bR4[j];
        }
    };
#endif
    // MODE 2: bf16-pair register staging with LDS transpose store
    u32 bR2[8];
    auto loadB2 = [&](int k0) {
#pragma unroll
        for (int j = 0; j < 8; ++j) {
            int idx = tid + j * 256;                 // 2048 bf16-pairs
            int kk = idx >> 6, nn2 = (idx & 63) << 1;
            bR2[j] = *(const u32*)(W + (size_t)(k0 + kk) * N + n0 + nn2);
        }
    };
    auto storeB2 = [&]() {
#pragma unroll
        for (int j = 0; j < 8; ++j) {
            int idx = tid + j * 256;
            int kk = idx >> 6, nn2 = (idx & 63) << 1;
            const __bf16* pr = (const __bf16*)&bR2[j];
            Bsd[(nn2    ) * SA + kk] = pr[0];
            Bsd[(nn2 + 1) * SA + kk] = pr[1];
        }
    };

    int p = 0;
    loadA(kBeg);
    if (MODE == 2) loadB2(kBeg);
#if HAVE_ASYNC_LDS
    else issueB(kBeg, 0);
#else
    else loadB1(kBeg);
#endif

    for (int k0 = kBeg; k0 < kEnd; k0 += BK) {
        storeA();
        if (MODE == 2) storeB2();
#if HAVE_ASYNC_LDS
        if (MODE == 1) ASYNC_WAIT();      // buffer p landed (this wave's copies)
#else
        if (MODE == 1) storeB1();
#endif
        __syncthreads();
        if (k0 + BK < kEnd) {             // overlap tile k+1 transfer with WMMA
            loadA(k0 + BK);
            if (MODE == 2) loadB2(k0 + BK);
#if HAVE_ASYNC_LDS
            else issueB(k0 + BK, p ^ 1);
#else
            else loadB1(k0 + BK);
#endif
        }
        const __bf16* BsBase = &Bsd[(MODE == 1) ? p * BN * SA : 0];

        // fragment loads: ISA 16-bit striping k(i) = (i&7)+((i>>3)<<4)+half*8
        // -> two aligned 16-byte LDS reads per fragment
        bf16x16 afr[2], bfr[2];
#pragma unroll
        for (int t = 0; t < 2; ++t) {
            const __bf16* ap = &As[(wm + t * 16 + l16) * SA + half * 8];
            bf16x8 alo = *(const bf16x8*)ap;
            bf16x8 ahi = *(const bf16x8*)(ap + 16);
            afr[t] = __builtin_shufflevector(alo, ahi, 0,1,2,3,4,5,6,7,8,9,10,11,12,13,14,15);
            const __bf16* bp = &BsBase[(wn + t * 16 + l16) * SA + half * 8];
            bf16x8 blo = *(const bf16x8*)bp;
            bf16x8 bhi = *(const bf16x8*)(bp + 16);
            bfr[t] = __builtin_shufflevector(blo, bhi, 0,1,2,3,4,5,6,7,8,9,10,11,12,13,14,15);
        }
#pragma unroll
        for (int tm = 0; tm < 2; ++tm)
#pragma unroll
            for (int tn = 0; tn < 2; ++tn)
                acc[tm][tn] = __builtin_amdgcn_wmma_f32_16x16x32_bf16(
                    false, afr[tm], false, bfr[tn], (short)0, acc[tm][tn], false, false);
        __syncthreads();
        if (MODE == 1 && HAVE_ASYNC_LDS) p ^= 1;
    }

    // epilogue: C/D layout = VGPR j, lane -> row = j + half*8, col = lane&15
    const bool split = (gridDim.z > 1);
#pragma unroll
    for (int tm = 0; tm < 2; ++tm) {
#pragma unroll
        for (int tn = 0; tn < 2; ++tn) {
            const int col  = n0 + wn + tn * 16 + l16;
            const int rowb = m0 + wm + tm * 16 + half * 8;
#pragma unroll
            for (int j = 0; j < 8; ++j) {
                const int row = rowb + j;
                float v = acc[tm][tn][j];
                size_t ci = (size_t)row * N + col;
                if (split) {
                    atomicAdd(&C[ci], v);       // split-K partials (C pre-zeroed)
                } else {
                    if (bias)     v += bias[col];
                    if (residual) v += residual[ci];
                    C[ci] = v;
                }
            }
        }
    }
}

// ---------------------------------------------------------------------------
// Elementwise / small kernels
// ---------------------------------------------------------------------------
__global__ void copy_k(float* __restrict__ d, const float* __restrict__ s, int n)
{ int i = blockIdx.x * 256 + threadIdx.x; if (i < n) d[i] = s[i]; }

__global__ void zero_k(float* __restrict__ d, int n)
{ int i = blockIdx.x * 256 + threadIdx.x; if (i < n) d[i] = 0.0f; }

__global__ void cvt_bf16_k(__bf16* __restrict__ d, const float* __restrict__ s, int n)
{ int i = blockIdx.x * 256 + threadIdx.x; if (i < n) d[i] = (__bf16)s[i]; }

// transposed bf16 mirror: d[n*K+k] = bf16(s[k*N+n]); writes coalesced
__global__ void cvt_bf16_t_k(__bf16* __restrict__ d, const float* __restrict__ s,
                             int K, int N)
{
    size_t i = (size_t)blockIdx.x * 256 + threadIdx.x;
    if (i >= (size_t)K * N) return;
    int n = (int)(i / K), k = (int)(i % K);
    d[i] = (__bf16)s[(size_t)k * N + n];
}

// SGD for biases (fp32 master only)
__global__ void sgd_k(float* __restrict__ p, const float* __restrict__ g, int n)
{
    int i = blockIdx.x * 256 + threadIdx.x; if (i >= n) return;
    p[i] -= LRc * g[i];
}

// SGD for weight matrices: fp32 master + row-major bf16 + transposed bf16
__global__ void sgd_wk(float* __restrict__ p, const float* __restrict__ g,
                       __bf16* __restrict__ pb, __bf16* __restrict__ pt,
                       int rows, int cols)
{
    int i = blockIdx.x * 256 + threadIdx.x;
    if (i >= rows * cols) return;
    float v = p[i] - LRc * g[i];
    p[i] = v;
    __bf16 h = (__bf16)v;
    pb[i] = h;
    int k = i / cols, n = i % cols;
    pt[(size_t)n * rows + k] = h;
}

__global__ void embed_k(const int* __restrict__ x, const float* __restrict__ emb,
                        const float* __restrict__ pos, float* __restrict__ h0, int c)
{
    int i = blockIdx.x * 256 + threadIdx.x;
    if (i >= Nrow * Dm) return;
    int r = i / Dm, d = i % Dm;
    int b = r / MBm, p = r % MBm;
    int tok = x[b * Tm + c * MBm + p];
    h0[i] = emb[(size_t)tok * Dm + d] + pos[p * Dm + d];
}

__global__ void gelu_fwd_k(const float* __restrict__ p, float* __restrict__ o, int n)
{
    int i = blockIdx.x * 256 + threadIdx.x; if (i >= n) return;
    float x = p[i];
    float u = 0.7978845608028654f * (x + 0.044715f * x * x * x);
    o[i] = 0.5f * x * (1.0f + tanhf(u));
}

__global__ void gelu_bwd_k(const float* __restrict__ dg, const float* __restrict__ p,
                           float* __restrict__ dp, __bf16* __restrict__ dpb, int n)
{
    int i = blockIdx.x * 256 + threadIdx.x; if (i >= n) return;
    float x = p[i];
    float u = 0.7978845608028654f * (x + 0.044715f * x * x * x);
    float t = tanhf(u);
    float d = 0.5f * (1.0f + t)
            + 0.5f * x * (1.0f - t * t) * 0.7978845608028654f * (1.0f + 3.0f * 0.044715f * x * x);
    float v = dg[i] * d;
    dp[i] = v;
    dpb[i] = (__bf16)v;
}

__global__ void colsum_k(const float* __restrict__ X, float* __restrict__ o,
                         int rows, int cols)
{
    int j = blockIdx.x * 256 + threadIdx.x; if (j >= cols) return;
    float s = 0.0f;
    for (int m = 0; m < rows; ++m) s += X[(size_t)m * cols + j];
    o[j] = s;
}

// ---------------------------------------------------------------------------
// LayerNorm fwd/bwd (one block per row, D=512)
// ---------------------------------------------------------------------------
__global__ __launch_bounds__(128)
void ln_fwd(const float* __restrict__ x, const float* __restrict__ sc,
            const float* __restrict__ bi, float* __restrict__ y,
            float* __restrict__ mean, float* __restrict__ rstd)
{
    const int row = blockIdx.x, tid = threadIdx.x;
    const float* xr = x + (size_t)row * Dm;
    __shared__ float red[128];
    float s = 0.0f;
    for (int i = tid; i < Dm; i += 128) s += xr[i];
    red[tid] = s; __syncthreads();
    for (int o = 64; o; o >>= 1) { if (tid < o) red[tid] += red[tid + o]; __syncthreads(); }
    float mu = red[0] / (float)Dm; __syncthreads();
    float v = 0.0f;
    for (int i = tid; i < Dm; i += 128) { float d = xr[i] - mu; v += d * d; }
    red[tid] = v; __syncthreads();
    for (int o = 64; o; o >>= 1) { if (tid < o) red[tid] += red[tid + o]; __syncthreads(); }
    float r = rsqrtf(red[0] / (float)Dm + 1e-5f);
    for (int i = tid; i < Dm; i += 128)
        y[(size_t)row * Dm + i] = (xr[i] - mu) * r * sc[i] + bi[i];
    if (tid == 0) { mean[row] = mu; rstd[row] = r; }
}

// dx = r*(dy*s - mean(dy*s) - xhat*mean(dy*s*xhat)) [+ base]; optional bf16 mirror
__global__ __launch_bounds__(128)
void ln_bwd(const float* __restrict__ dy, const float* __restrict__ x,
            const float* __restrict__ mean, const float* __restrict__ rstd,
            const float* __restrict__ sc, const float* __restrict__ base,
            float* __restrict__ dx, __bf16* __restrict__ dxb)
{
    const int row = blockIdx.x, tid = threadIdx.x;
    const float* xr = x + (size_t)row * Dm;
    const float* dyr = dy + (size_t)row * Dm;
    __shared__ float r1[128], r2[128];
    const float mu = mean[row], r = rstd[row];
    float s1 = 0.0f, s2 = 0.0f;
    for (int i = tid; i < Dm; i += 128) {
        float g = dyr[i] * sc[i];
        float xh = (xr[i] - mu) * r;
        s1 += g; s2 += g * xh;
    }
    r1[tid] = s1; r2[tid] = s2; __syncthreads();
    for (int o = 64; o; o >>= 1) {
        if (tid < o) { r1[tid] += r1[tid + o]; r2[tid] += r2[tid + o]; }
        __syncthreads();
    }
    s1 = r1[0] / (float)Dm; s2 = r2[0] / (float)Dm;
    for (int i = tid; i < Dm; i += 128) {
        float g = dyr[i] * sc[i];
        float xh = (xr[i] - mu) * r;
        float v = r * (g - s1 - xh * s2) + (base ? base[(size_t)row * Dm + i] : 0.0f);
        dx[(size_t)row * Dm + i] = v;
        if (dxb) dxb[(size_t)row * Dm + i] = (__bf16)v;
    }
}

// ---------------------------------------------------------------------------
// Attention: S=128, DH=64, one block per (b,h); whole head staged in LDS via
// CDNA5 async global->LDS copies (320KB WGP LDS). Online softmax fwd; stats
// recomputed in bwd (no stored probability matrix).
// Row stride 68 floats keeps every 16B async segment aligned.
// ---------------------------------------------------------------------------
__global__ __launch_bounds__(128)
void attn_fwd(const float* __restrict__ qkv, float* __restrict__ oatt)
{
    __shared__ __attribute__((aligned(16))) float qs[MBm][DHm + 4];
    __shared__ __attribute__((aligned(16))) float ks[MBm][DHm + 4];
    __shared__ __attribute__((aligned(16))) float vs[MBm][DHm + 4];
    const int bh = blockIdx.x, b = bh / NHm, hh = bh % NHm;
    const int s = threadIdx.x;
    const int co = hh * DHm;
#if HAVE_ASYNC_LDS
    for (int it = 0; it < 48; ++it) {                 // 6144 16B segments
        int sgi = s + it * 128;
        int mat = sgi >> 11;                          // 0:q 1:k 2:v
        int rem = sgi & 2047;
        int row = rem >> 4, seg = rem & 15;
        const float* gp = qkv + (size_t)(b * MBm + row) * H3 + mat * Dm + co + seg * 4;
        float* lp = (mat == 0) ? &qs[row][seg * 4]
                  : (mat == 1) ? &ks[row][seg * 4] : &vs[row][seg * 4];
        __builtin_amdgcn_global_load_async_to_lds_b128(AS_GLOBAL(gp), AS_LDS(lp), 0, 0);
    }
    ASYNC_WAIT();
#else
    {
        const size_t base = (size_t)(b * MBm + s) * H3;
        for (int d = 0; d < DHm; ++d) {
            qs[s][d] = qkv[base + co + d];
            ks[s][d] = qkv[base + Dm + co + d];
            vs[s][d] = qkv[base + 2 * Dm + co + d];
        }
    }
#endif
    __syncthreads();
    float m = -3.4e38f, l = 0.0f;
    float o[DHm];
#pragma unroll
    for (int d = 0; d < DHm; ++d) o[d] = 0.0f;
    for (int j = 0; j <= s; ++j) {
        float scv = 0.0f;
#pragma unroll
        for (int d = 0; d < DHm; ++d) scv += qs[s][d] * ks[j][d];
        scv *= 0.125f;
        float mn = fmaxf(m, scv);
        float corr = __expf(m - mn);
        float pe = __expf(scv - mn);
        l = l * corr + pe;
#pragma unroll
        for (int d = 0; d < DHm; ++d) o[d] = o[d] * corr + pe * vs[j][d];
        m = mn;
    }
    float inv = 1.0f / l;
#pragma unroll 8
    for (int d = 0; d < DHm; ++d)
        oatt[(size_t)(b * MBm + s) * Dm + co + d] = o[d] * inv;
}

__global__ __launch_bounds__(128)
void attn_bwd(const float* __restrict__ qkv, const float* __restrict__ dov,
              float* __restrict__ dqkv)
{
    __shared__ __attribute__((aligned(16))) float qs[MBm][DHm + 4];
    __shared__ __attribute__((aligned(16))) float ks[MBm][DHm + 4];
    __shared__ __attribute__((aligned(16))) float vs[MBm][DHm + 4];
    __shared__ __attribute__((aligned(16))) float dos[MBm][DHm + 4];
    __shared__ float ms[MBm], ls[MBm], dels[MBm];
    const int bh = blockIdx.x, b = bh / NHm, hh = bh % NHm;
    const int t = threadIdx.x;
    const size_t rb = (size_t)(b * MBm + t);
    const size_t base = rb * H3;
    const int co = hh * DHm;
#if HAVE_ASYNC_LDS
    for (int it = 0; it < 64; ++it) {                 // 8192 16B segments
        int sgi = t + it * 128;
        int mat = sgi >> 11;                          // 0:q 1:k 2:v 3:do
        int rem = sgi & 2047;
        int row = rem >> 4, seg = rem & 15;
        const float* gp = (mat < 3)
            ? qkv + (size_t)(b * MBm + row) * H3 + mat * Dm + co + seg * 4
            : dov + (size_t)(b * MBm + row) * Dm + co + seg * 4;
        float* lp = (mat == 0) ? &qs[row][seg * 4]
                  : (mat == 1) ? &ks[row][seg * 4]
                  : (mat == 2) ? &vs[row][seg * 4] : &dos[row][seg * 4];
        __builtin_amdgcn_global_load_async_to_lds_b128(AS_GLOBAL(gp), AS_LDS(lp), 0, 0);
    }
    ASYNC_WAIT();
#else
    for (int d = 0; d < DHm; ++d) {
        qs[t][d]  = qkv[base + co + d];
        ks[t][d]  = qkv[base + Dm + co + d];
        vs[t][d]  = qkv[base + 2 * Dm + co + d];
        dos[t][d] = dov[rb * Dm + co + d];
    }
#endif
    __syncthreads();
    // Phase A: softmax stats for row t
    {
        float m = -3.4e38f, l = 0.0f;
        for (int j = 0; j <= t; ++j) {
            float scv = 0.0f;
#pragma unroll
            for (int d = 0; d < DHm; ++d) scv += qs[t][d] * ks[j][d];
            scv *= 0.125f;
            float mn = fmaxf(m, scv);
            l = l * __expf(m - mn) + __expf(scv - mn);
            m = mn;
        }
        ms[t] = m; ls[t] = l;
    }
    __syncthreads();
    // Phase B: delta_t and dq row t
    {
        const float m = ms[t], invl = 1.0f / ls[t];
        float delta = 0.0f;
        for (int j = 0; j <= t; ++j) {
            float scv = 0.0f, dp = 0.0f;
#pragma unroll
            for (int d = 0; d < DHm; ++d) { scv += qs[t][d] * ks[j][d]; dp += dos[t][d] * vs[j][d]; }
            float pe = __expf(scv * 0.125f - m) * invl;
            delta += pe * dp;
        }
        dels[t] = delta;
        float dq[DHm];
#pragma unroll
        for (int d = 0; d < DHm; ++d) dq[d] = 0.0f;
        for (int j = 0; j <= t; ++j) {
            float scv = 0.0f, dp = 0.0f;
#pragma unroll
            for (int d = 0; d < DHm; ++d) { scv += qs[t][d] * ks[j][d]; dp += dos[t][d] * vs[j][d]; }
            float pe = __expf(scv * 0.125f - m) * invl;
            float dsv = pe * (dp - delta);
#pragma unroll
            for (int d = 0; d < DHm; ++d) dq[d] += dsv * ks[j][d];
        }
#pragma unroll 8
        for (int d = 0; d < DHm; ++d) dqkv[base + co + d] = dq[d] * 0.125f;
    }
    __syncthreads();
    // Phase C: dk, dv for column t
    {
        float dk[DHm], dv[DHm];
#pragma unroll
        for (int d = 0; d < DHm; ++d) { dk[d] = 0.0f; dv[d] = 0.0f; }
        for (int sI = t; sI < MBm; ++sI) {
            float scv = 0.0f, dp = 0.0f;
#pragma unroll
            for (int d = 0; d < DHm; ++d) { scv += qs[sI][d] * ks[t][d]; dp += dos[sI][d] * vs[t][d]; }
            float pe = __expf(scv * 0.125f - ms[sI]) / ls[sI];
            float dsv = pe * (dp - dels[sI]);
#pragma unroll
            for (int d = 0; d < DHm; ++d) { dv[d] += pe * dos[sI][d]; dk[d] += dsv * qs[sI][d]; }
        }
#pragma unroll 8
        for (int d = 0; d < DHm; ++d) {
            dqkv[base + Dm + co + d]     = dk[d] * 0.125f;
            dqkv[base + 2 * Dm + co + d] = dv[d];
        }
    }
}

// ---------------------------------------------------------------------------
// Cross-entropy grad: dlogits = (softmax(logits) - onehot(tgt)) / (B*MB)
// Logits are read straight from d_out (head GEMM wrote them there).
// ---------------------------------------------------------------------------
__global__ __launch_bounds__(256)
void loss_grad(const float* __restrict__ logits_out, const int* __restrict__ x,
               float* __restrict__ dlog, int c)
{
    const int r = blockIdx.x, tid = threadIdx.x;
    const int b = r / MBm, p = r % MBm;
    const float* lr = logits_out + ((size_t)b * (Tm - 1) + (size_t)c * MBm + p) * Vm;
    const int tgt = x[b * Tm + c * MBm + p + 1];
    __shared__ float red[256];
    float mx = -3.4e38f;
    for (int v = tid; v < Vm; v += 256) mx = fmaxf(mx, lr[v]);
    red[tid] = mx; __syncthreads();
    for (int o = 128; o; o >>= 1) { if (tid < o) red[tid] = fmaxf(red[tid], red[tid + o]); __syncthreads(); }
    mx = red[0]; __syncthreads();
    float s = 0.0f;
    for (int v = tid; v < Vm; v += 256) s += __expf(lr[v] - mx);
    red[tid] = s; __syncthreads();
    for (int o = 128; o; o >>= 1) { if (tid < o) red[tid] += red[tid + o]; __syncthreads(); }
    const float inv = 1.0f / red[0];
    const float scl = 1.0f / (float)Nrow;
    float* dr = dlog + (size_t)r * Vm;
    for (int v = tid; v < Vm; v += 256) {
        float pv = __expf(lr[v] - mx) * inv;
        dr[v] = (pv - (v == tgt ? 1.0f : 0.0f)) * scl;
    }
}

// ---------------------------------------------------------------------------
// Host orchestration
// ---------------------------------------------------------------------------
extern "C" void kernel_launch(void* const* d_in, const int* in_sizes, int n_in,
                              void* d_out, int out_size, void* d_ws, size_t ws_size,
                              hipStream_t stream)
{
    (void)in_sizes; (void)n_in; (void)out_size; (void)ws_size;
    const int*   x     = (const int*)  d_in[0];
    const float* emb   = (const float*)d_in[1];
    const float* pos   = (const float*)d_in[2];
    const float* ln1s  = (const float*)d_in[3];
    const float* ln1b  = (const float*)d_in[4];
    const float* Wqkv  = (const float*)d_in[5];
    const float* bqkv  = (const float*)d_in[6];
    const float* Wo    = (const float*)d_in[7];
    const float* bo    = (const float*)d_in[8];
    const float* ln2s  = (const float*)d_in[9];
    const float* ln2b  = (const float*)d_in[10];
    const float* W1    = (const float*)d_in[11];
    const float* b1    = (const float*)d_in[12];
    const float* W2    = (const float*)d_in[13];
    const float* b2    = (const float*)d_in[14];
    const float* lnfs  = (const float*)d_in[15];
    const float* lnfb  = (const float*)d_in[16];
    const float* Whead = (const float*)d_in[17];
    const float* bhead = (const float*)d_in[18];
    float* out = (float*)d_out;
    float* ws  = (float*)d_ws;

    // workspace layout (float units, 32B aligned); ~233 MB total
    size_t off = 0;
    auto alloc = [&](size_t n) { size_t o = off; off += (n + 7) & ~(size_t)7; return o; };
    const size_t pW1  = alloc((size_t)Lm * Dm * HIDm);
    const size_t pb1  = alloc((size_t)Lm * HIDm);
    const size_t pW2  = alloc((size_t)Lm * HIDm * Dm);
    const size_t pb2  = alloc((size_t)Lm * Dm);
    const size_t h0   = alloc((size_t)(Lm + 1) * Nrow * Dm);
    const size_t qkvs = alloc((size_t)Lm * Nrow * H3);
    const size_t hmid = alloc((size_t)Lm * Nrow * Dm);
    const size_t mbuf = alloc((size_t)Lm * Nrow * Dm);
    const size_t pre1 = alloc((size_t)Lm * Nrow * HIDm);
    const size_t gel  = alloc((size_t)Lm * Nrow * HIDm);
    const size_t ln1m = alloc((size_t)Lm * Nrow);
    const size_t ln1r = alloc((size_t)Lm * Nrow);
    const size_t ln2m = alloc((size_t)Lm * Nrow);
    const size_t ln2r = alloc((size_t)Lm * Nrow);
    const size_t lnfm = alloc(Nrow);
    const size_t lnfr = alloc(Nrow);
    const size_t a1   = alloc((size_t)Nrow * Dm);
    const size_t oatt = alloc((size_t)Nrow * Dm);
    const size_t lnfo = alloc((size_t)Nrow * Dm);
    const size_t dlog = alloc((size_t)Nrow * Vm);
    const size_t gA   = alloc((size_t)Nrow * Dm);
    const size_t gB   = alloc((size_t)Nrow * Dm);
    const size_t dhm  = alloc((size_t)Nrow * Dm);
    const size_t dmb  = alloc((size_t)Nrow * Dm);
    const size_t dob  = alloc((size_t)Nrow * Dm);
    const size_t dab  = alloc((size_t)Nrow * Dm);
    const size_t dgel = alloc((size_t)Nrow * HIDm);
    const size_t dpre = alloc((size_t)Nrow * HIDm);
    const size_t dqkb = alloc((size_t)Nrow * H3);
    const size_t dW1b = alloc((size_t)Dm * HIDm);
    const size_t db1b = alloc(HIDm);
    const size_t dW2b = alloc((size_t)HIDm * Dm);
    const size_t db2b = alloc(Dm);
    // bf16 mirrors (sizes in fp32 units = n/2): row-major (backward NT) and
    // transposed (forward-as-NT)
    const size_t bWqkv = alloc((size_t)Lm * Dm * H3 / 2);
    const size_t bWo   = alloc((size_t)Lm * Dm * Dm / 2);
    const size_t bWhd  = alloc((size_t)Dm * Vm / 2);
    const size_t bW1   = alloc((size_t)Lm * Dm * HIDm / 2);
    const size_t bW2   = alloc((size_t)Lm * HIDm * Dm / 2);
    const size_t tWqkv = alloc((size_t)Lm * Dm * H3 / 2);
    const size_t tWo   = alloc((size_t)Lm * Dm * Dm / 2);
    const size_t tWhd  = alloc((size_t)Dm * Vm / 2);
    const size_t tW1   = alloc((size_t)Lm * Dm * HIDm / 2);
    const size_t tW2   = alloc((size_t)Lm * HIDm * Dm / 2);
    const size_t bgA   = alloc((size_t)Nrow * Dm / 2);
    const size_t bgB   = alloc((size_t)Nrow * Dm / 2);
    const size_t bdpr  = alloc((size_t)Nrow * HIDm / 2);

    __bf16* BWqkv = (__bf16*)(ws + bWqkv);
    __bf16* BWo   = (__bf16*)(ws + bWo);
    __bf16* BWhd  = (__bf16*)(ws + bWhd);
    __bf16* BW1   = (__bf16*)(ws + bW1);
    __bf16* BW2   = (__bf16*)(ws + bW2);
    __bf16* TWqkv = (__bf16*)(ws + tWqkv);
    __bf16* TWo   = (__bf16*)(ws + tWo);
    __bf16* TWhd  = (__bf16*)(ws + tWhd);
    __bf16* TW1   = (__bf16*)(ws + tW1);
    __bf16* TW2   = (__bf16*)(ws + tW2);
    __bf16* BgA   = (__bf16*)(ws + bgA);
    __bf16* BgB   = (__bf16*)(ws + bgB);
    __bf16* Bdpr  = (__bf16*)(ws + bdpr);

    auto ew = [](size_t n) { return dim3((unsigned)((n + 255) / 256)); };

    // forward GEMM: C = A @ W (W pre-transposed to [N][K] -> NT kernel)
    auto gemm_fwd = [&](const float* A, const __bf16* Wt, float* Cp, int M, int Nn, int Kk,
                        const float* bias, const float* res) {
        dim3 g(Nn / BN, M / BM, 1);
        gemm_wmma<1><<<g, 256, 0, stream>>>(A, Wt, Cp, M, Nn, Kk, bias, res, Kk);
    };
    auto gemm_nt = [&](const float* A, const __bf16* Wm, float* Cp, int M, int Nn, int Kk,
                       int splitZ) {
        dim3 g(Nn / BN, M / BM, 1);
        int kc = Kk;
        if (splitZ > 1) {
            kc = ((Kk / splitZ + BK - 1) / BK) * BK;
            g.z = (Kk + kc - 1) / kc;
            zero_k<<<ew((size_t)M * Nn), 256, 0, stream>>>(Cp, M * Nn);
        }
        gemm_wmma<1><<<g, 256, 0, stream>>>(A, Wm, Cp, M, Nn, Kk, nullptr, nullptr, kc);
    };
    auto gemm_tn = [&](const float* A, const __bf16* Bp, float* Cp, int M, int Nn, int Kk) {
        dim3 g(Nn / BN, M / BM, 1);
        gemm_wmma<2><<<g, 256, 0, stream>>>(A, Bp, Cp, M, Nn, Kk, nullptr, nullptr, Kk);
    };

    // fresh fp32 masters + bf16 mirrors (row-major and transposed) every call
    copy_k<<<ew((size_t)Lm * Dm * HIDm), 256, 0, stream>>>(ws + pW1, W1, Lm * Dm * HIDm);
    copy_k<<<ew((size_t)Lm * HIDm),      256, 0, stream>>>(ws + pb1, b1, Lm * HIDm);
    copy_k<<<ew((size_t)Lm * HIDm * Dm), 256, 0, stream>>>(ws + pW2, W2, Lm * HIDm * Dm);
    copy_k<<<ew((size_t)Lm * Dm),        256, 0, stream>>>(ws + pb2, b2, Lm * Dm);
    cvt_bf16_k<<<ew((size_t)Lm * Dm * HIDm), 256, 0, stream>>>(BW1, W1, Lm * Dm * HIDm);
    cvt_bf16_k<<<ew((size_t)Lm * HIDm * Dm), 256, 0, stream>>>(BW2, W2, Lm * HIDm * Dm);
    cvt_bf16_k<<<ew((size_t)Lm * Dm * H3),   256, 0, stream>>>(BWqkv, Wqkv, Lm * Dm * H3);
    cvt_bf16_k<<<ew((size_t)Lm * Dm * Dm),   256, 0, stream>>>(BWo, Wo, Lm * Dm * Dm);
    cvt_bf16_k<<<ew((size_t)Dm * Vm),        256, 0, stream>>>(BWhd, Whead, Dm * Vm);
    for (int l = 0; l < Lm; ++l) {
        cvt_bf16_t_k<<<ew((size_t)Dm * H3), 256, 0, stream>>>(TWqkv + (size_t)l * Dm * H3,
                                                              Wqkv + (size_t)l * Dm * H3, Dm, H3);
        cvt_bf16_t_k<<<ew((size_t)Dm * Dm), 256, 0, stream>>>(TWo + (size_t)l * Dm * Dm,
                                                              Wo + (size_t)l * Dm * Dm, Dm, Dm);
        cvt_bf16_t_k<<<ew((size_t)Dm * HIDm), 256, 0, stream>>>(TW1 + (size_t)l * Dm * HIDm,
                                                                W1 + (size_t)l * Dm * HIDm, Dm, HIDm);
        cvt_bf16_t_k<<<ew((size_t)HIDm * Dm), 256, 0, stream>>>(TW2 + (size_t)l * HIDm * Dm,
                                                                W2 + (size_t)l * HIDm * Dm, HIDm, Dm);
    }
    cvt_bf16_t_k<<<ew((size_t)Dm * Vm), 256, 0, stream>>>(TWhd, Whead, Dm, Vm);

    const int nchunks = (Tm - 1) / MBm;  // 8
    for (int c = 0; c < nchunks; ++c) {
        // ---------------- forward ----------------
        embed_k<<<ew((size_t)Nrow * Dm), 256, 0, stream>>>(x, emb, pos, ws + h0, c);
        for (int l = 0; l < Lm; ++l) {
            float* hl   = ws + h0 + (size_t)l * Nrow * Dm;
            float* hl1  = hl + (size_t)Nrow * Dm;
            float* qkvl = ws + qkvs + (size_t)l * Nrow * H3;
            float* hmdl = ws + hmid + (size_t)l * Nrow * Dm;
            float* ml   = ws + mbuf + (size_t)l * Nrow * Dm;
            float* prel = ws + pre1 + (size_t)l * Nrow * HIDm;
            float* gll  = ws + gel  + (size_t)l * Nrow * HIDm;
            ln_fwd<<<Nrow, 128, 0, stream>>>(hl, ln1s + l * Dm, ln1b + l * Dm,
                                             ws + a1, ws + ln1m + l * Nrow, ws + ln1r + l * Nrow);
            gemm_fwd(ws + a1, TWqkv + (size_t)l * Dm * H3, qkvl, Nrow, H3, Dm, bqkv + l * H3, nullptr);
            attn_fwd<<<Bm * NHm, 128, 0, stream>>>(qkvl, ws + oatt);
            gemm_fwd(ws + oatt, TWo + (size_t)l * Dm * Dm, hmdl, Nrow, Dm, Dm, bo + l * Dm, hl);
            ln_fwd<<<Nrow, 128, 0, stream>>>(hmdl, ln2s + l * Dm, ln2b + l * Dm,
                                             ml, ws + ln2m + l * Nrow, ws + ln2r + l * Nrow);
            gemm_fwd(ml, TW1 + (size_t)l * Dm * HIDm, prel, Nrow, HIDm, Dm,
                     ws + pb1 + (size_t)l * HIDm, nullptr);
            gelu_fwd_k<<<ew((size_t)Nrow * HIDm), 256, 0, stream>>>(prel, gll, Nrow * HIDm);
            gemm_fwd(gll, TW2 + (size_t)l * HIDm * Dm, hl1, Nrow, Dm, HIDm,
                     ws + pb2 + (size_t)l * Dm, hmdl);
        }
        float* hL = ws + h0 + (size_t)Lm * Nrow * Dm;
        ln_fwd<<<Nrow, 128, 0, stream>>>(hL, lnfs, lnfb, ws + lnfo, ws + lnfm, ws + lnfr);
        for (int b = 0; b < Bm; ++b)  // head GEMM writes logits directly into d_out
            gemm_fwd(ws + lnfo + (size_t)b * MBm * Dm, TWhd,
                     out + ((size_t)b * (Tm - 1) + (size_t)c * MBm) * Vm,
                     MBm, Vm, Dm, bhead, nullptr);
        loss_grad<<<Nrow, 256, 0, stream>>>(out, x, ws + dlog, c);

        // ---------------- backward ----------------
        gemm_nt(ws + dlog, BWhd, ws + dmb, Nrow, Dm, Vm, 16);   // split-K dh
        ln_bwd<<<Nrow, 128, 0, stream>>>(ws + dmb, hL, ws + lnfm, ws + lnfr, lnfs,
                                         nullptr, ws + gA, BgA);
        float* g = ws + gA;  __bf16* gb = BgA;
        float* gn = ws + gB; __bf16* gnb = BgB;
        for (int l = Lm - 1; l >= 0; --l) {
            float* hl   = ws + h0 + (size_t)l * Nrow * Dm;
            float* qkvl = ws + qkvs + (size_t)l * Nrow * H3;
            float* hmdl = ws + hmid + (size_t)l * Nrow * Dm;
            float* ml   = ws + mbuf + (size_t)l * Nrow * Dm;
            float* prel = ws + pre1 + (size_t)l * Nrow * HIDm;
            float* gll  = ws + gel  + (size_t)l * Nrow * HIDm;
            // MLP branch (reads old params; updated after use)
            gemm_nt(g, BW2 + (size_t)l * HIDm * Dm, ws + dgel, Nrow, HIDm, Dm, 1);
            gemm_tn(gll, gb, ws + dW2b, HIDm, Dm, Nrow);
            colsum_k<<<ew(Dm), 256, 0, stream>>>(g, ws + db2b, Nrow, Dm);
            gelu_bwd_k<<<ew((size_t)Nrow * HIDm), 256, 0, stream>>>(ws + dgel, prel,
                                                                    ws + dpre, Bdpr, Nrow * HIDm);
            gemm_tn(ml, Bdpr, ws + dW1b, Dm, HIDm, Nrow);
            colsum_k<<<ew(HIDm), 256, 0, stream>>>(ws + dpre, ws + db1b, Nrow, HIDm);
            gemm_nt(ws + dpre, BW1 + (size_t)l * Dm * HIDm, ws + dmb, Nrow, Dm, HIDm, 1);
            ln_bwd<<<Nrow, 128, 0, stream>>>(ws + dmb, hmdl, ws + ln2m + l * Nrow,
                                             ws + ln2r + l * Nrow, ln2s + l * Dm, g,
                                             ws + dhm, (__bf16*)nullptr);
            // SGD update: fp32 master + row-major and transposed bf16 mirrors
            sgd_wk<<<ew((size_t)Dm * HIDm), 256, 0, stream>>>(ws + pW1 + (size_t)l * Dm * HIDm,
                                                              ws + dW1b, BW1 + (size_t)l * Dm * HIDm,
                                                              TW1 + (size_t)l * Dm * HIDm, Dm, HIDm);
            sgd_k<<<ew(HIDm), 256, 0, stream>>>(ws + pb1 + (size_t)l * HIDm, ws + db1b, HIDm);
            sgd_wk<<<ew((size_t)HIDm * Dm), 256, 0, stream>>>(ws + pW2 + (size_t)l * HIDm * Dm,
                                                              ws + dW2b, BW2 + (size_t)l * HIDm * Dm,
                                                              TW2 + (size_t)l * HIDm * Dm, HIDm, Dm);
            sgd_k<<<ew(Dm), 256, 0, stream>>>(ws + pb2 + (size_t)l * Dm, ws + db2b, Dm);
            // attention branch
            gemm_nt(ws + dhm, BWo + (size_t)l * Dm * Dm, ws + dob, Nrow, Dm, Dm, 1);
            attn_bwd<<<Bm * NHm, 128, 0, stream>>>(qkvl, ws + dob, ws + dqkb);
            gemm_nt(ws + dqkb, BWqkv + (size_t)l * Dm * H3, ws + dab, Nrow, Dm, H3, 1);
            ln_bwd<<<Nrow, 128, 0, stream>>>(ws + dab, hl, ws + ln1m + l * Nrow,
                                             ws + ln1r + l * Nrow, ln1s + l * Dm, ws + dhm,
                                             gn, gnb);
            float* tf = g; g = gn; gn = tf;
            __bf16* tb = gb; gb = gnb; gnb = tb;
        }
    }
}
